// CrossPixContrastive_41678362640816
// MI455X (gfx1250) — matrix operand order repaired and checked by reading, not compiled, not run on hardware
//
#include <hip/hip_runtime.h>
#include <hip/hip_bf16.h>

// CDNA5 / gfx1250: wave32, WMMA bf16 16x16x32 (f32 accum), async global->LDS.
typedef __bf16 bf16_t;
typedef __attribute__((ext_vector_type(16))) __bf16 v16bf;
typedef __attribute__((ext_vector_type(8)))  __bf16 v8bf;
typedef __attribute__((ext_vector_type(8)))  float  v8f;

#define N_   4
#define C_   256
#define HW_  4096
// exp(x/0.05) = exp2(x * 20/ln2)
#define EXP_SCALE 28.853900817779268f
#define LDSTRIDE 72   // 64 + 8 bf16 pad -> 36 dword row stride, conflict-free b128 lane pattern
#define ASZ (256 * LDSTRIDE)   // one A buffer, bf16 elems
#define BSZ (128 * LDSTRIDE)   // one B buffer, bf16 elems

// gfx1250 async global->LDS copy, 16B per lane, tracked by ASYNCcnt.
__device__ __forceinline__ void async_ld_b128(unsigned lds_byte_off, const bf16_t* gaddr) {
  asm volatile("global_load_async_to_lds_b128 %0, %1, off"
               :: "v"(lds_byte_off), "v"((unsigned long long)(uintptr_t)gaddr)
               : "memory");
}
// local->flat addrspacecast is {aperture_hi32, lds_offset32}: truncate for VDST.
__device__ __forceinline__ unsigned lds_off(const void* p) {
  return (unsigned)(uintptr_t)p;
}

// ---------------------------------------------------------------------------
// Kernel 0: zero the reduction buffers (rowE/rowNum/colE/colNum + accum).
// ---------------------------------------------------------------------------
__global__ void zero_kernel(float* __restrict__ p, int count) {
  int i = blockIdx.x * blockDim.x + threadIdx.x;
  if (i < count) p[i] = 0.0f;
}

// ---------------------------------------------------------------------------
// Kernel 1: per-pixel L2 normalize over C, convert to bf16, transpose from
// (N,C,H,W) to pixel-major (N,HW,C). Thread <-> pixel; reads coalesced.
// ---------------------------------------------------------------------------
__global__ void norm_bf16_kernel(const float* __restrict__ rgb,
                                 const float* __restrict__ ir,
                                 bf16_t* __restrict__ Rn,
                                 bf16_t* __restrict__ In) {
  const float* src = (blockIdx.y == 0) ? rgb : ir;
  bf16_t*      dst = (blockIdx.y == 0) ? Rn  : In;
  int n = blockIdx.x >> 4;
  int p = ((blockIdx.x & 15) << 8) + threadIdx.x;
  const float* base = src + (size_t)n * C_ * HW_ + p;

  float ss = 0.0f;
  for (int c = 0; c < C_; ++c) {
    float x = base[(size_t)c * HW_];
    ss += x * x;
  }
  float scale = 1.0f / fmaxf(sqrtf(ss), 1e-12f);

  bf16_t* out = dst + ((size_t)n * HW_ + p) * C_;
  for (int c = 0; c < C_; c += 8) {
    v8bf chunk;
#pragma unroll
    for (int j = 0; j < 8; ++j)
      chunk[j] = (bf16_t)(base[(size_t)(c + j) * HW_] * scale);
    *(v8bf*)(out + c) = chunk;
  }
}

// ---------------------------------------------------------------------------
// Kernel 2: fused GEMM + contrastive epilogue.
// Grid (16,32,4): 256x128 tile of the 4096x4096 similarity matrix per block.
// 8 waves (4x2): wave tile 64x64 = 4x4 WMMA accumulators (16x16 f32).
// K staged in 64-wide chunks via DOUBLE-BUFFERED LDS filled by
// global_load_async_to_lds_b128 (ASYNCcnt), overlapping with WMMA compute.
// Epilogue: clamp -> exp2 -> label-match, shuffle-reduce, global f32 atomics.
// ---------------------------------------------------------------------------
__global__ __launch_bounds__(256)
void contrast_gemm_kernel(const bf16_t* __restrict__ Rn,
                          const bf16_t* __restrict__ In,
                          const int* __restrict__ rgb_mask,
                          const int* __restrict__ ir_mask,
                          float* __restrict__ rowE, float* __restrict__ rowNum,
                          float* __restrict__ colE, float* __restrict__ colNum) {
  __shared__ bf16_t Alds[2 * ASZ];
  __shared__ bf16_t Blds[2 * BSZ];
  __shared__ int rmL[256];
  __shared__ int imL[128];

  const int n       = blockIdx.z;
  const int rowBase = blockIdx.x * 256;
  const int colBase = blockIdx.y * 128;
  const int tid  = threadIdx.x;
  const int lane = tid & 31;
  const int wave = tid >> 5;
  const int wr   = wave >> 1;      // 0..3 -> 64-row strip
  const int wc   = wave & 1;       // 0..1 -> 64-col strip
  const int j16  = lane & 15;
  const int h    = lane >> 4;

  rmL[tid] = rgb_mask[n * HW_ + rowBase + tid];
  if (tid < 128) imL[tid] = ir_mask[n * HW_ + colBase + tid];

  const bf16_t* Abase = Rn + ((size_t)n * HW_ + rowBase) * C_;
  const bf16_t* Bbase = In + ((size_t)n * HW_ + colBase) * C_;

  const int lr = tid >> 3;          // 0..31 : row within a 32-row load pass
  const int lk = (tid & 7) * 8;     // 0..56 : 16B-aligned k offset (64-wide chunk)

  // Issue one K-chunk's async loads: A 256x64 (8 passes) + B 128x64 (4 passes).
  auto stage_async = [&](int kb, int buf) {
    bf16_t* Ab = Alds + buf * ASZ;
    bf16_t* Bb = Blds + buf * BSZ;
#pragma unroll
    for (int pass = 0; pass < 8; ++pass) {
      int r = pass * 32 + lr;
      async_ld_b128(lds_off(Ab + r * LDSTRIDE + lk),
                    Abase + (size_t)r * C_ + kb + lk);
      if (pass < 4)
        async_ld_b128(lds_off(Bb + r * LDSTRIDE + lk),
                      Bbase + (size_t)r * C_ + kb + lk);
    }
  };

  v8f zero8 = {0.f, 0.f, 0.f, 0.f, 0.f, 0.f, 0.f, 0.f};
  v8f acc[4][4];
#pragma unroll
  for (int tr = 0; tr < 4; ++tr)
#pragma unroll
    for (int tc = 0; tc < 4; ++tc) acc[tr][tc] = zero8;

  stage_async(0, 0);                       // prologue: fill buffer 0

  for (int kb = 0; kb < C_; kb += 64) {
    const int buf = (kb >> 6) & 1;
    __syncthreads();                       // all reads of buf^1 (prev chunk) done
    if (kb + 64 < C_) {
      stage_async(kb + 64, buf ^ 1);       // overlap next chunk with this compute
      asm volatile("s_wait_asynccnt 0xc" ::: "memory");  // 12 newest in flight
    } else {
      asm volatile("s_wait_asynccnt 0x0" ::: "memory");
    }
    __syncthreads();                       // current buf visible to all waves

    const bf16_t* Ab = Alds + buf * ASZ;
    const bf16_t* Bb = Blds + buf * BSZ;
#pragma unroll
    for (int ks = 0; ks < 64; ks += 32) {
      v16bf afrag[4];
      v16bf bfrag[4];
      // A fragment (16x32 bf16): lane (m=j16,h): K = ks+8h+0..7 , ks+16+8h+0..7
#pragma unroll
      for (int tr = 0; tr < 4; ++tr) {
        int row = wr * 64 + tr * 16 + j16;
        const bf16_t* pA = Ab + row * LDSTRIDE + ks + 8 * h;
        v8bf lo = *(const v8bf*)pA;
        v8bf hi = *(const v8bf*)(pA + 16);
        afrag[tr] = __builtin_shufflevector(lo, hi, 0,1,2,3,4,5,6,7,8,9,10,11,12,13,14,15);
      }
      // B fragment (32x16 bf16): lane (n=j16,h): K = ks+8h+0..7 , ks+16+8h+0..7
#pragma unroll
      for (int tc = 0; tc < 4; ++tc) {
        int col = wc * 64 + tc * 16 + j16;
        const bf16_t* pB = Bb + col * LDSTRIDE + ks + 8 * h;
        v8bf lo = *(const v8bf*)pB;
        v8bf hi = *(const v8bf*)(pB + 16);
        bfrag[tc] = __builtin_shufflevector(lo, hi, 0,1,2,3,4,5,6,7,8,9,10,11,12,13,14,15);
      }
#pragma unroll
      for (int tr = 0; tr < 4; ++tr)
#pragma unroll
        for (int tc = 0; tc < 4; ++tc)
          acc[tr][tc] = __builtin_amdgcn_wmma_f32_16x16x32_bf16(
              false, afrag[tr], false, bfrag[tc], (short)0, acc[tr][tc],
              false, false);
    }
  }

  // ---- fused epilogue: e = exp(clamp(s)/T); row & col sums (all + matched) ----
  float colS[4] = {0.f, 0.f, 0.f, 0.f};
  float colN[4] = {0.f, 0.f, 0.f, 0.f};
  int myIm[4];
#pragma unroll
  for (int tc = 0; tc < 4; ++tc) myIm[tc] = imL[wc * 64 + tc * 16 + j16];

#pragma unroll
  for (int tr = 0; tr < 4; ++tr) {
#pragma unroll
    for (int v = 0; v < 8; ++v) {
      int lrow = wr * 64 + tr * 16 + v + 8 * h;   // C/D layout: row = v + 8h
      int rmv  = rmL[lrow];
      float rS = 0.f, rN = 0.f;
#pragma unroll
      for (int tc = 0; tc < 4; ++tc) {
        float s = acc[tr][tc][v];
        s = fminf(1.0f, fmaxf(-1.0f, s));
        float e = exp2f(s * EXP_SCALE);
        float num = (rmv == myIm[tc]) ? e : 0.f;
        rS += e;  rN += num;
        colS[tc] += e;  colN[tc] += num;
      }
      // reduce across the 16 columns held by this lane's half
#pragma unroll
      for (int m = 1; m < 16; m <<= 1) {
        rS += __shfl_xor(rS, m, 32);
        rN += __shfl_xor(rN, m, 32);
      }
      if (j16 == 0) {
        atomicAdd(&rowE  [n * HW_ + rowBase + lrow], rS);
        atomicAdd(&rowNum[n * HW_ + rowBase + lrow], rN);
      }
    }
  }
#pragma unroll
  for (int tc = 0; tc < 4; ++tc) {
    // lanes j and j+16 hold the same column (different row halves)
    colS[tc] += __shfl_xor(colS[tc], 16, 32);
    colN[tc] += __shfl_xor(colN[tc], 16, 32);
    if (h == 0) {
      int lcol = wc * 64 + tc * 16 + j16;
      atomicAdd(&colE  [n * HW_ + colBase + lcol], colS[tc]);
      atomicAdd(&colNum[n * HW_ + colBase + lcol], colN[tc]);
    }
  }
}

// ---------------------------------------------------------------------------
// Kernel 3: loss = concat(rowNum/rowE+eps, colNum/colE+eps) * fg;
// sum of -log over nonzero entries + count, via block reduce + 2 atomics.
// ---------------------------------------------------------------------------
__global__ void loss_kernel(const float* __restrict__ rowE,
                            const float* __restrict__ rowNum,
                            const float* __restrict__ colE,
                            const float* __restrict__ colNum,
                            const int* __restrict__ rgb_mask,
                            const int* __restrict__ ir_mask,
                            float* __restrict__ accum) {
  __shared__ float sT[256];
  __shared__ float sC[256];
  int idx  = blockIdx.x * 256 + threadIdx.x;   // 0..2*N*HW-1 = 32767
  int side = idx >> 14;                        // N*HW = 16384
  int i    = idx & 16383;

  int   fg  = ((side == 0) ? rgb_mask[i] : ir_mask[i]) > 0;
  float num = (side == 0) ? rowNum[i] : colNum[i];
  float den = (side == 0) ? rowE[i]   : colE[i];
  float r   = num / (den + 1e-6f);
  float val = fg ? r : 0.f;
  bool  nz  = (val != 0.f);

  sT[threadIdx.x] = nz ? -logf(val) : 0.f;
  sC[threadIdx.x] = nz ? 1.f : 0.f;
  __syncthreads();
  for (int s = 128; s > 0; s >>= 1) {
    if (threadIdx.x < s) {
      sT[threadIdx.x] += sT[threadIdx.x + s];
      sC[threadIdx.x] += sC[threadIdx.x + s];
    }
    __syncthreads();
  }
  if (threadIdx.x == 0) {
    atomicAdd(&accum[0], sT[0]);
    atomicAdd(&accum[1], sC[0]);
  }
}

__global__ void finalize_kernel(const float* __restrict__ accum,
                                float* __restrict__ out) {
  out[0] = accum[0] / fmaxf(accum[1], 1.0f);
}

// ---------------------------------------------------------------------------
extern "C" void kernel_launch(void* const* d_in, const int* in_sizes, int n_in,
                              void* d_out, int out_size, void* d_ws, size_t ws_size,
                              hipStream_t stream) {
  const float* rgb = (const float*)d_in[0];
  const float* ir  = (const float*)d_in[1];
  const int*   rm  = (const int*)d_in[2];
  const int*   im  = (const int*)d_in[3];

  char* ws = (char*)d_ws;
  // workspace layout (~16.3 MB total)
  bf16_t* Rn   = (bf16_t*)ws;                              // 8 MB
  bf16_t* In   = (bf16_t*)(ws + (size_t)8388608);          // 8 MB
  float* rowE   = (float*)(ws + (size_t)16777216);         // 64 KB each
  float* rowNum = rowE   + N_ * HW_;
  float* colE   = rowNum + N_ * HW_;
  float* colNum = colE   + N_ * HW_;
  float* accum  = colNum + N_ * HW_;                       // 2 floats

  int zcount = 4 * N_ * HW_ + 2;
  zero_kernel<<<(zcount + 255) / 256, 256, 0, stream>>>(rowE, zcount);

  norm_bf16_kernel<<<dim3(64, 2), 256, 0, stream>>>(rgb, ir, Rn, In);

  contrast_gemm_kernel<<<dim3(16, 32, N_), 256, 0, stream>>>(
      Rn, In, rm, im, rowE, rowNum, colE, colNum);

  loss_kernel<<<(2 * N_ * HW_) / 256, 256, 0, stream>>>(
      rowE, rowNum, colE, colNum, rm, im, accum);

  finalize_kernel<<<1, 1, 0, stream>>>(accum, (float*)d_out);
}
